// Model_78451872628838
// MI455X (gfx1250) — matrix operand order repaired and checked by reading, not compiled, not run on hardware
//
#include <hip/hip_runtime.h>
#include <hip/hip_bf16.h>
#include <math.h>

typedef __attribute__((ext_vector_type(16))) __bf16 v16bf;
typedef __attribute__((ext_vector_type(8)))  float  v8f;
typedef __attribute__((ext_vector_type(4)))  unsigned int v4u;

// ---------------------------------------------------------------------------
// WMMA: D = A(16x32 bf16) * B(32x16 bf16) + C(f32)
// ---------------------------------------------------------------------------
__device__ __forceinline__ v8f wmma_bf16(v16bf a, v16bf b, v8f c) {
  return __builtin_amdgcn_wmma_f32_16x16x32_bf16(
      false, a, false, b, (short)0, c, false, false);
}

// ---------------------------------------------------------------------------
// Fragment loaders. Tiles in LDS are row-major [rows][ldk] bf16
// (row = M for A, row = N for B; all GEMMs are "NT").
// A 16x32 frag: lanes 0-15 -> K 0..7 & 16..23 of row M=lane;
//               lanes 16-31 -> K 8..15 & 24..31.
// B 32x16 frag: lane holds column N=lane&15; lanes 0-15 K 0..7 & 16..23,
//               lanes 16-31 K 8..15 & 24..31.
// ---------------------------------------------------------------------------
template <int LDK>
__device__ __forceinline__ v16bf load_a_frag(const __bf16* tile, int rowBase,
                                             int kc, int lane) {
  const __bf16* p = tile + (rowBase + (lane & 15)) * LDK + kc;
  int kb = (lane < 16) ? 0 : 8;
  union { v16bf v; v4u u[2]; } r;
  r.u[0] = *(const v4u*)(p + kb);
  r.u[1] = *(const v4u*)(p + kb + 16);
  return r.v;
}

template <int LDK>
__device__ __forceinline__ v16bf load_b_frag(const __bf16* tile, int colBase,
                                             int kc, int lane) {
  const __bf16* p = tile + (colBase + (lane & 15)) * LDK + kc;
  int kb = (lane < 16) ? 0 : 8;
  union { v16bf v; v4u u[2]; } r;
  r.u[0] = *(const v4u*)(p + kb);
  r.u[1] = *(const v4u*)(p + kb + 16);
  return r.v;
}

// ---------------------------------------------------------------------------
// Staging helpers.
// ---------------------------------------------------------------------------
// 64x32 f32 tile -> LDS bf16 (used by the big parallel GEMM whose sources
// are f32 inputs). 128 threads, thread t: 16 floats of row t/2.
__device__ __forceinline__ void stage_f32_64x32(__bf16* dst, const float* src,
                                                long rowStride, int tid) {
  int r = tid >> 1;
  int c = (tid & 1) << 4;
  const float* s = src + (size_t)r * (size_t)rowStride + c;
  float4 f0 = ((const float4*)s)[0];
  float4 f1 = ((const float4*)s)[1];
  float4 f2 = ((const float4*)s)[2];
  float4 f3 = ((const float4*)s)[3];
  __bf16* d = dst + r * 32 + c;
  d[0]=(__bf16)f0.x; d[1]=(__bf16)f0.y; d[2]=(__bf16)f0.z; d[3]=(__bf16)f0.w;
  d[4]=(__bf16)f1.x; d[5]=(__bf16)f1.y; d[6]=(__bf16)f1.z; d[7]=(__bf16)f1.w;
  d[8]=(__bf16)f2.x; d[9]=(__bf16)f2.y; d[10]=(__bf16)f2.z; d[11]=(__bf16)f2.w;
  d[12]=(__bf16)f3.x; d[13]=(__bf16)f3.y; d[14]=(__bf16)f3.z; d[15]=(__bf16)f3.w;
}

// 64x64 bf16 tile -> LDS bf16 straight copy (b128 in / b128 out).
// 128 threads, thread t: 32 halves (64B) of row t/2.
__device__ __forceinline__ void stage_bf16_64x64(__bf16* dst, const __bf16* src,
                                                 long rowStride, int tid) {
  int r = tid >> 1;
  int c = (tid & 1) << 5;
  const v4u* s = (const v4u*)(src + (size_t)r * (size_t)rowStride + c);
  v4u a0 = s[0], a1 = s[1], a2 = s[2], a3 = s[3];
  v4u* d = (v4u*)(dst + r * 64 + c);
  d[0] = a0; d[1] = a1; d[2] = a2; d[3] = a3;
}

__device__ __forceinline__ void stage_zero_64x64(__bf16* dst, int tid) {
  int r = tid >> 1;
  int c = (tid & 1) << 5;
  v4u z = {0u, 0u, 0u, 0u};
  v4u* d = (v4u*)(dst + r * 64 + c);
  d[0] = z; d[1] = z; d[2] = z; d[3] = z;
}

// ---------------------------------------------------------------------------
// Kernel 1: C[m,n] = sum_k A[m,k]*Bw[n,k] + bias[n]  (f32 sources, cvt at
// staging). Writes C in f32 AND bf16 (second copy feeds the LSTM x path).
// Grid (M/64, N/64), 128 threads, wave = 32x32 quadrant, K-chunk 32.
// ---------------------------------------------------------------------------
__global__ __launch_bounds__(128) void gemm_nt_bias_dual(
    const float* __restrict__ A, long aStride,
    const float* __restrict__ Bw, long bStride,
    const float* __restrict__ bias,
    float* __restrict__ C, __bf16* __restrict__ Cbf, long cStride, int K) {
  __shared__ __align__(16) __bf16 As[64 * 32];
  __shared__ __align__(16) __bf16 Bs[64 * 32];
  int tid = threadIdx.x;
  int lane = tid & 31, wave = tid >> 5;
  int wm = (wave >> 1) * 32;
  int wn = (wave & 1) * 32;
  long mBase = (long)blockIdx.x * 64;
  long nBase = (long)blockIdx.y * 64;

  v8f acc[2][2] = {{{}, {}}, {{}, {}}};
  for (int k0 = 0; k0 < K; k0 += 32) {
    __syncthreads();
    stage_f32_64x32(As, A + (size_t)mBase * aStride + k0, aStride, tid);
    stage_f32_64x32(Bs, Bw + (size_t)nBase * bStride + k0, bStride, tid);
    __syncthreads();
    v16bf a0 = load_a_frag<32>(As, wm, 0, lane);
    v16bf a1 = load_a_frag<32>(As, wm + 16, 0, lane);
    v16bf b0 = load_b_frag<32>(Bs, wn, 0, lane);
    v16bf b1 = load_b_frag<32>(Bs, wn + 16, 0, lane);
    acc[0][0] = wmma_bf16(a0, b0, acc[0][0]);
    acc[0][1] = wmma_bf16(a0, b1, acc[0][1]);
    acc[1][0] = wmma_bf16(a1, b0, acc[1][0]);
    acc[1][1] = wmma_bf16(a1, b1, acc[1][1]);
  }
  int rsel = (lane < 16) ? 0 : 8;
#pragma unroll
  for (int ns = 0; ns < 2; ++ns) {
    long col = nBase + wn + ns * 16 + (lane & 15);
    float bv = bias[col];
#pragma unroll
    for (int ms = 0; ms < 2; ++ms) {
#pragma unroll
      for (int v = 0; v < 8; ++v) {
        long row = mBase + wm + ms * 16 + rsel + v;
        float val = acc[ms][ns][v] + bv;
        C[(size_t)row * cStride + col] = val;
        Cbf[(size_t)row * cStride + col] = (__bf16)val;
      }
    }
  }
}

// ---------------------------------------------------------------------------
// Kernel 2: one LSTM timestep gate GEMM, pure bf16 sources.
// G[m,n] = x[m,:]·Wg[n,0:512] + h[m,:]·Wg[n,512:1024] + bsum[n]
// M=128, N=2048, K=1024; Wg is fused [W_ih | W_hh] bf16 (2048x1024).
// K-chunk 64 (8 WMMA per barrier pair) to shorten the dependent chain.
// Grid (2, 32), 128 threads.
// ---------------------------------------------------------------------------
__global__ __launch_bounds__(128) void lstm_gate_gemm_bf(
    const __bf16* __restrict__ Xbf, long xStride,
    const __bf16* __restrict__ Hbf,        // 128x512 bf16, null -> zeros (t==0)
    const __bf16* __restrict__ Wg,         // 2048x1024
    const float* __restrict__ bsum,        // 2048
    float* __restrict__ G) {
  __shared__ __align__(16) __bf16 As[64 * 64];
  __shared__ __align__(16) __bf16 Bs[64 * 64];
  int tid = threadIdx.x;
  int lane = tid & 31, wave = tid >> 5;
  int wm = (wave >> 1) * 32;
  int wn = (wave & 1) * 32;
  long mBase = (long)blockIdx.x * 64;
  long nBase = (long)blockIdx.y * 64;

  v8f acc[2][2] = {{{}, {}}, {{}, {}}};
  for (int k0 = 0; k0 < 1024; k0 += 64) {
    __syncthreads();
    if (k0 < 512)
      stage_bf16_64x64(As, Xbf + (size_t)mBase * xStride + k0, xStride, tid);
    else if (Hbf)
      stage_bf16_64x64(As, Hbf + (size_t)mBase * 512 + (k0 - 512), 512, tid);
    else
      stage_zero_64x64(As, tid);
    stage_bf16_64x64(Bs, Wg + (size_t)nBase * 1024 + k0, 1024, tid);
    __syncthreads();
#pragma unroll
    for (int kc = 0; kc < 64; kc += 32) {
      v16bf a0 = load_a_frag<64>(As, wm, kc, lane);
      v16bf a1 = load_a_frag<64>(As, wm + 16, kc, lane);
      v16bf b0 = load_b_frag<64>(Bs, wn, kc, lane);
      v16bf b1 = load_b_frag<64>(Bs, wn + 16, kc, lane);
      acc[0][0] = wmma_bf16(a0, b0, acc[0][0]);
      acc[0][1] = wmma_bf16(a0, b1, acc[0][1]);
      acc[1][0] = wmma_bf16(a1, b0, acc[1][0]);
      acc[1][1] = wmma_bf16(a1, b1, acc[1][1]);
    }
  }
  int rsel = (lane < 16) ? 0 : 8;
#pragma unroll
  for (int ns = 0; ns < 2; ++ns) {
    long col = nBase + wn + ns * 16 + (lane & 15);
    float bv = bsum[col];
#pragma unroll
    for (int ms = 0; ms < 2; ++ms) {
#pragma unroll
      for (int v = 0; v < 8; ++v) {
        long row = mBase + wm + ms * 16 + rsel + v;
        G[(size_t)row * 2048 + col] = acc[ms][ns][v] + bv;
      }
    }
  }
}

// ---------------------------------------------------------------------------
// Kernel 3: LSTM pointwise. Writes h (f32 -> output), h (bf16 -> next step's
// GEMM A-operand), c (f32 -> cs).
// ---------------------------------------------------------------------------
__device__ __forceinline__ float sigmoidf_(float x) {
  return 1.f / (1.f + __expf(-x));
}

__global__ __launch_bounds__(256) void lstm_pointwise(
    const float* __restrict__ G,
    const float* __restrict__ cprev, long cpStride,
    float* __restrict__ hdst, long hStride,
    float* __restrict__ cdst, long cStride,
    __bf16* __restrict__ hbf,
    int Bn, int Hn) {
  int idx = blockIdx.x * blockDim.x + threadIdx.x;
  if (idx >= Bn * Hn) return;
  int m = idx / Hn, h = idx % Hn;
  const float* g = G + (size_t)m * 4 * Hn;
  float gi = g[h];
  float gf = g[Hn + h];
  float gc = g[2 * Hn + h];
  float go = g[3 * Hn + h];
  float c = cprev ? cprev[(size_t)m * cpStride + h] : 0.f;
  float c2 = sigmoidf_(gf) * c + sigmoidf_(gi) * tanhf(gc);
  float h2 = sigmoidf_(go) * tanhf(c2);
  hdst[(size_t)m * hStride + h] = h2;
  cdst[(size_t)m * cStride + h] = c2;
  hbf[(size_t)m * Hn + h] = (__bf16)h2;
}

// ---------------------------------------------------------------------------
// Kernel 4: build fused bf16 weight matrix Wg = [W_ih | W_hh] and bsum.
// ---------------------------------------------------------------------------
__global__ __launch_bounds__(256) void build_wg(
    const float* __restrict__ Wih, const float* __restrict__ Whh,
    const float* __restrict__ bih, const float* __restrict__ bhh,
    __bf16* __restrict__ Wg, float* __restrict__ bsum) {
  int idx = blockIdx.x * 256 + threadIdx.x;   // 2048*1024 total
  int n = idx >> 10, k = idx & 1023;
  float v = (k < 512) ? Wih[(size_t)n * 512 + k]
                      : Whh[(size_t)n * 512 + (k - 512)];
  Wg[idx] = (__bf16)v;
  if (idx < 2048) bsum[idx] = bih[idx] + bhh[idx];
}

// ---------------------------------------------------------------------------
// Kernel 5: last-row attention scores a_last[b,s] = q_b · values[b,s].
// ---------------------------------------------------------------------------
__global__ __launch_bounds__(128) void attn_scores(
    const float* __restrict__ vatt, const float* __restrict__ values,
    float* __restrict__ a_last, int Bn, int Tn, int Dn) {
  int w = blockIdx.x * 4 + (threadIdx.x >> 5);
  int lane = threadIdx.x & 31;
  int S = Tn - 1;
  if (w >= Bn * S) return;
  int b = w / S, s = w % S;
  const float* q = vatt + ((size_t)b * Tn + (Tn - 1)) * Dn;
  const float* v = values + ((size_t)b * Tn + s) * Dn;
  float sum = 0.f;
  for (int k = lane; k < Dn; k += 32) sum += q[k] * v[k];
#pragma unroll
  for (int off = 16; off > 0; off >>= 1) sum += __shfl_down(sum, off);
  if (lane == 0) a_last[(size_t)b * S + s] = sum;
}

// ---------------------------------------------------------------------------
// Kernel 6: row softmax over S scores. One block per batch row.
// ---------------------------------------------------------------------------
__global__ __launch_bounds__(128) void softmax_rows(
    const float* __restrict__ a, float* __restrict__ alpha, int S) {
  __shared__ float red[128];
  int b = blockIdx.x, tid = threadIdx.x;
  const float* row = a + (size_t)b * S;
  float m = -1e30f;
  for (int i = tid; i < S; i += 128) m = fmaxf(m, row[i]);
  red[tid] = m; __syncthreads();
  for (int st = 64; st > 0; st >>= 1) {
    if (tid < st) red[tid] = fmaxf(red[tid], red[tid + st]);
    __syncthreads();
  }
  m = red[0]; __syncthreads();
  float sum = 0.f;
  for (int i = tid; i < S; i += 128) sum += __expf(row[i] - m);
  red[tid] = sum; __syncthreads();
  for (int st = 64; st > 0; st >>= 1) {
    if (tid < st) red[tid] += red[tid + st];
    __syncthreads();
  }
  float inv = 1.f / red[0];
  for (int i = tid; i < S; i += 128)
    alpha[(size_t)b * S + i] = __expf(row[i] - m) * inv;
}

// ---------------------------------------------------------------------------
// Kernel 7: C_i[b,h] = sum_j alpha[b,j]*cs[b,j,h]/log(e + Del[b,j+1,h]),
// Del = flip(cumsum(Deltas)). Streaming: walk s ascending keeping the
// cumsum; reverse index j = S-1-s. Grid (B, H/128).
// ---------------------------------------------------------------------------
__global__ __launch_bounds__(128) void ci_reduce(
    const float* __restrict__ Deltas, const float* __restrict__ cs,
    const float* __restrict__ alpha, float* __restrict__ Ci,
    int Tn, int Dn) {
  __shared__ float alds[512];
  int b = blockIdx.x;
  int h = blockIdx.y * 128 + threadIdx.x;
  int S = Tn - 1;
  for (int i = threadIdx.x; i < S; i += 128)
    alds[i] = alpha[(size_t)b * S + i];
  __syncthreads();
  const float* dptr = Deltas + (size_t)b * Tn * Dn + h;
  const float* cptr = cs + (size_t)b * S * Dn + h;
  float cum = 0.f, acc = 0.f;
  for (int s = 0; s < S; ++s) {
    cum += dptr[(size_t)s * Dn];
    int j = S - 1 - s;
    acc += alds[j] * cptr[(size_t)j * Dn] / __logf(2.7182818284590452f + cum);
  }
  Ci[(size_t)b * Dn + h] = acc;
}

// ---------------------------------------------------------------------------
// Host-side launch sequence.
// ---------------------------------------------------------------------------
extern "C" void kernel_launch(void* const* d_in, const int* in_sizes, int n_in,
                              void* d_out, int out_size, void* d_ws, size_t ws_size,
                              hipStream_t stream) {
  (void)in_sizes; (void)n_in; (void)out_size; (void)ws_size;
  const float* values = (const float*)d_in[0];   // (B,T,D)
  const float* Deltas = (const float*)d_in[1];   // (B,T,D)
  const float* W_att  = (const float*)d_in[2];   // (D,D)
  const float* b_att  = (const float*)d_in[3];   // (D)
  const float* W_ih   = (const float*)d_in[4];   // (4H,D)
  const float* W_hh   = (const float*)d_in[5];   // (4H,H)
  const float* b_ih   = (const float*)d_in[6];   // (4H)
  const float* b_hh   = (const float*)d_in[7];   // (4H)
  float* out = (float*)d_out;                    // (B,T,H)

  const int B = 128, T = 512, D = 512, H = 512, S = T - 1;

  // Bump allocator over d_ws (256B-aligned chunks).
  char* base = (char*)d_ws;
  size_t off = 0;
  auto alloc = [&](size_t bytes) -> char* {
    off = (off + 255) & ~(size_t)255;
    char* p = base + off;
    off += bytes;
    return p;
  };
  float*  v_att   = (float*)alloc((size_t)B * T * D * 4);      // 134 MB
  __bf16* vatt_bf = (__bf16*)alloc((size_t)B * T * D * 2);     // 67 MB
  float*  cs      = (float*)alloc((size_t)B * S * H * 4);      // 134 MB
  float*  G       = (float*)alloc((size_t)B * 4 * H * 4);      // 1 MB
  __bf16* Wg      = (__bf16*)alloc((size_t)2048 * 1024 * 2);   // 4 MB
  float*  bsum    = (float*)alloc(2048 * 4);
  __bf16* hbf     = (__bf16*)alloc((size_t)B * H * 2);         // 128 KB
  float*  a_last  = (float*)alloc((size_t)B * S * 4);
  float*  alpha   = (float*)alloc((size_t)B * S * 4);
  float*  Ci      = (float*)alloc((size_t)B * H * 4);
  float*  cfin    = (float*)alloc((size_t)B * H * 4);

  // 0) fused bf16 weights + combined bias (once per call; lives in L2 after)
  build_wg<<<(2048 * 1024) / 256, 256, 0, stream>>>(
      W_ih, W_hh, b_ih, b_hh, Wg, bsum);

  // 1) v_att = values @ W_att.T + b_att  (dual f32 + bf16 output)
  gemm_nt_bias_dual<<<dim3((B * T) / 64, D / 64), 128, 0, stream>>>(
      values, D, W_att, D, b_att, v_att, vatt_bf, D, D);

  // 2) last-row attention scores + softmax
  attn_scores<<<(B * S + 3) / 4, 128, 0, stream>>>(v_att, values, a_last, B, T, D);
  softmax_rows<<<B, 128, 0, stream>>>(a_last, alpha, S);

  // 3) sequential LSTM scan, t = 0..T-2 (hs written straight into d_out)
  for (int t = 0; t < S; ++t) {
    const __bf16* Hp = (t == 0) ? nullptr : hbf;
    const float*  cp = (t == 0) ? nullptr : cs + (size_t)(t - 1) * H;
    lstm_gate_gemm_bf<<<dim3(2, 32), 128, 0, stream>>>(
        vatt_bf + (size_t)t * D, (long)T * D, Hp, Wg, bsum, G);
    lstm_pointwise<<<(B * H + 255) / 256, 256, 0, stream>>>(
        G, cp, (long)S * H,
        out + (size_t)t * H, (long)T * H,
        cs + (size_t)t * H, (long)S * H,
        hbf, B, H);
  }

  // 4) C_i reduction (streaming cumsum of Deltas; no Del materialization)
  ci_reduce<<<dim3(B, H / 128), 128, 0, stream>>>(Deltas, cs, alpha, Ci, T, D);

  // 5) final cell: x = v_att[:,T-1], h = hs[:,T-2] (still in hbf), c = C_i
  lstm_gate_gemm_bf<<<dim3(2, 32), 128, 0, stream>>>(
      vatt_bf + (size_t)(T - 1) * D, (long)T * D, hbf, Wg, bsum, G);
  lstm_pointwise<<<(B * H + 255) / 256, 256, 0, stream>>>(
      G, Ci, (long)H,
      out + (size_t)(T - 1) * H, (long)T * H,
      cfin, (long)H,
      hbf, B, H);
}
